// GraphEncoder_9062380994867
// MI455X (gfx1250) — matrix operand order repaired and checked by reading, not compiled, not run on hardware
//
#include <hip/hip_runtime.h>
#include <hip/hip_bf16.h>

// ---------------------------------------------------------------------------
// Transformer encoder forward for gfx1250 (MI455X).
// All matmul operands are bf16 in global memory; GEMM tiles are staged with
// CDNA5 async global->LDS DMA (ASYNCcnt) and consumed by v_wmma_f32_16x16x32_bf16.
// L=3, D=256, H=8, HD=32, FF=1024, B=32, S=512, M = B*S = 16384.
// ---------------------------------------------------------------------------

#define LAYERS 3
#define DMODEL 256
#define NHEADS 8
#define HDIM   32
#define FFDIM  1024
#define SEQ    512
#define BATCH  32
#define MTOT   (BATCH * SEQ)          // 16384

typedef __attribute__((ext_vector_type(16))) __bf16 v16bf;
typedef __attribute__((ext_vector_type(8)))  float  v8f;

// ---- CDNA5 async global->LDS copy (16B per lane), tracked by ASYNCcnt -----
__device__ inline void async_b128(const void* gsrc, unsigned lds_off) {
  asm volatile("global_load_async_to_lds_b128 %0, %1, off"
               :: "v"(lds_off), "v"(gsrc) : "memory");
}
template <int N>
__device__ inline void wait_async() {
  asm volatile("s_wait_asynccnt %0" :: "i"(N) : "memory");
}
__device__ inline unsigned lds_off_of(const void* p) {
  // generic pointers to LDS carry the 32-bit LDS byte offset in the low half
  return (unsigned)(unsigned long long)(uintptr_t)p;
}

// ---- WMMA fragment loaders (contiguous k order, same for A and B) -----------
// A 16x32 bf16: lane l holds row M=(l&15); element e -> K = 16*(l>>4)+e
__device__ inline v16bf load_a_frag(const __bf16* rowp, int half) {
  v16bf a;
#pragma unroll
  for (int e = 0; e < 16; ++e) {
    int kk = 16 * half + e;
    a[e] = rowp[kk];
  }
  return a;
}
// B 32x16 bf16: lane l holds col N=(l&15); element e -> K = 16*(l>>4)+e
// (B is staged [n][k] so this is one contiguous 16-half read)
__device__ inline v16bf load_b_frag(const __bf16* colp, int half) {
  v16bf b;
  const __bf16* p = colp + 16 * half;
#pragma unroll
  for (int e = 0; e < 16; ++e) b[e] = p[e];
  return b;
}

// ---------------------------------------------------------------------------
// GEMM: C[M,N] = A[M,K] (bf16) * Wt[N,K]^T (bf16) + bias[N]; optional ReLU;
// output either fp32 or bf16.  BM=128 x BN=64, BK=32, 256 thr = 8 waves,
// each wave a 32x32 subtile (2x2 WMMAs).  Double-buffered async staging.
// ---------------------------------------------------------------------------
#define BM 128
#define BN 64
#define BK 32
#define ALD (BK + 8)   // padded LDS stride in bf16 (80B rows -> conflict-free)

__global__ __launch_bounds__(256)
void gemm_bf16_async(const __bf16* __restrict__ A, const __bf16* __restrict__ Wt,
                     const float* __restrict__ bias, void* __restrict__ Cout,
                     int M, int N, int K, int relu, int out_bf) {
  __shared__ __bf16 As[2][BM * ALD];
  __shared__ __bf16 Bs[2][BN * ALD];

  const int tid  = threadIdx.x;
  const int m0   = blockIdx.y * BM;
  const int n0   = blockIdx.x * BN;
  const int wave = tid >> 5;
  const int lane = tid & 31;
  const int half = lane >> 4;
  const int l16  = lane & 15;
  const int wm   = wave & 3;   // rows 32*wm .. +32
  const int wn   = wave >> 2;  // cols 32*wn .. +32

  const unsigned asb = lds_off_of(&As[0][0]);
  const unsigned bsb = lds_off_of(&Bs[0][0]);

  auto issue_tile = [&](int buf, int k0) {
    // A tile: 128 rows x 32 k  = 512 x 16B chunks -> 2 per thread
#pragma unroll
    for (int p = 0; p < 2; ++p) {
      const int idx = p * 256 + tid;
      const int row = idx >> 2, c = idx & 3;
      async_b128(A + (size_t)(m0 + row) * K + k0 + c * 8,
                 asb + (unsigned)(buf * (BM * ALD * 2) + row * (ALD * 2) + c * 16));
    }
    // B tile: 64 rows([n]) x 32 k = 256 x 16B chunks -> 1 per thread
    {
      const int row = tid >> 2, c = tid & 3;
      async_b128(Wt + (size_t)(n0 + row) * K + k0 + c * 8,
                 bsb + (unsigned)(buf * (BN * ALD * 2) + row * (ALD * 2) + c * 16));
    }
  };

  v8f acc[2][2] = {};
  const int KT = K / BK;

  issue_tile(0, 0);
  for (int kt = 0; kt < KT; ++kt) {
    const int cur = kt & 1;
    if (kt + 1 < KT) {
      issue_tile(cur ^ 1, (kt + 1) * BK);
      wait_async<3>();          // 3 async ops of the *next* tile still in flight
    } else {
      wait_async<0>();
    }
    __syncthreads();            // this tile visible to all waves

    v16bf af[2], bfm[2];
#pragma unroll
    for (int mt = 0; mt < 2; ++mt)
      af[mt] = load_a_frag(&As[cur][(32 * wm + 16 * mt + l16) * ALD], half);
#pragma unroll
    for (int nt = 0; nt < 2; ++nt)
      bfm[nt] = load_b_frag(&Bs[cur][(32 * wn + 16 * nt + l16) * ALD], half);

#pragma unroll
    for (int mt = 0; mt < 2; ++mt)
#pragma unroll
      for (int nt = 0; nt < 2; ++nt)
        acc[mt][nt] = __builtin_amdgcn_wmma_f32_16x16x32_bf16(
            false, af[mt], false, bfm[nt], (short)0, acc[mt][nt], false, false);
    __syncthreads();            // done reading `cur` before it is re-filled
  }

  // epilogue: D layout: lane (l16) = N, VGPR r -> M = 8*half + r
#pragma unroll
  for (int mt = 0; mt < 2; ++mt) {
#pragma unroll
    for (int nt = 0; nt < 2; ++nt) {
      const int n = n0 + 32 * wn + 16 * nt + l16;
      const float bv = bias[n];
#pragma unroll
      for (int r = 0; r < 8; ++r) {
        const int m = m0 + 32 * wm + 16 * mt + 8 * half + r;
        float v = acc[mt][nt][r] + bv;
        if (relu) v = fmaxf(v, 0.f);
        if (out_bf) ((__bf16*)Cout)[(size_t)m * N + n] = (__bf16)v;
        else        ((float*)Cout)[(size_t)m * N + n] = v;
      }
    }
  }
}

// ---------------------------------------------------------------------------
// Fused attention: one block per (batch, head, 16-row query chunk).
// Q/K tiles DMA'd async (bf16, already K-contiguous); V transposed manually.
// scores: 1 WMMA K-step; masked scaled softmax; P@V over 16 K-steps.
// ---------------------------------------------------------------------------
#define QC 16
#define KLD 40            // bf16 stride for [s][hd] tiles (80B rows)
#define SLD (SEQ + 4)     // fp32 score stride
#define PLD (SEQ + 8)     // bf16 prob / V^T stride

__global__ __launch_bounds__(256)
void attn_kernel(const __bf16* __restrict__ Qb, const __bf16* __restrict__ Kb,
                 const __bf16* __restrict__ Vb, const unsigned char* __restrict__ mask,
                 __bf16* __restrict__ AOb) {
  const int q0 = blockIdx.x * QC;
  const int hh = blockIdx.y;
  const int bb = blockIdx.z;

  __shared__ __bf16 Qs[QC * KLD];
  __shared__ __bf16 Ks[SEQ * KLD];       // [s][hd]  (B-frag [n][k] for QK^T)
  __shared__ __bf16 Vts[HDIM * PLD];     // [hd][s]  (B-frag [n][k] for P@V)
  __shared__ float  Sc[QC * SLD];
  __shared__ __bf16 Pbf[QC * PLD];

  const int tid  = threadIdx.x;
  const int wave = tid >> 5;
  const int lane = tid & 31;
  const int half = lane >> 4;
  const int l16  = lane & 15;

  // ---- async stage K (512x32 bf16 = 2048 x 16B -> 8 per thread) ----
  {
    const unsigned ksb = lds_off_of(&Ks[0]);
#pragma unroll
    for (int p = 0; p < 8; ++p) {
      const int idx = p * 256 + tid;
      const int s = idx >> 2, c = idx & 3;
      async_b128(Kb + (size_t)(bb * SEQ + s) * DMODEL + hh * HDIM + c * 8,
                 ksb + (unsigned)(s * (KLD * 2) + c * 16));
    }
  }
  // ---- async stage Q (16x32 = 64 chunks, waves 0-1) ----
  if (tid < 64) {
    const unsigned qsb = lds_off_of(&Qs[0]);
    const int r = tid >> 2, c = tid & 3;
    async_b128(Qb + (size_t)(bb * SEQ + q0 + r) * DMODEL + hh * HDIM + c * 8,
               qsb + (unsigned)(r * (KLD * 2) + c * 16));
  }
  // ---- stage V transposed [hd][s] (manual: needs transpose) ----
  {
    const int rr = tid >> 3;          // 0..31
    const int c4 = (tid & 7) * 4;     // 0..28
#pragma unroll
    for (int p = 0; p < 16; ++p) {
      const int s = p * 32 + rr;
      const __bf16* g = Vb + (size_t)(bb * SEQ + s) * DMODEL + hh * HDIM + c4;
#pragma unroll
      for (int i = 0; i < 4; ++i) Vts[(c4 + i) * PLD + s] = g[i];
    }
  }
  wait_async<0>();
  __syncthreads();

  // ---- scores: each wave covers 64 key columns (4 N-tiles), M = 16 ----
  {
    const v16bf aq = load_a_frag(&Qs[l16 * KLD], half);
#pragma unroll
    for (int nt = 0; nt < 4; ++nt) {
      const int n = wave * 64 + nt * 16 + l16;
      const v16bf bk = load_b_frag(&Ks[n * KLD], half);
      v8f s = {};
      s = __builtin_amdgcn_wmma_f32_16x16x32_bf16(false, aq, false, bk,
                                                  (short)0, s, false, false);
#pragma unroll
      for (int r = 0; r < 8; ++r)
        Sc[(8 * half + r) * SLD + n] = s[r];
    }
  }
  __syncthreads();

  // ---- masked scaled softmax: 16 threads per row, 32 cols each ----
  {
    const float scale = 0.176776695296636881f;  // 1/sqrt(32)
    const int row = tid >> 4;
    const int sub = tid & 15;
    const unsigned char* mrow = mask + (size_t)bb * SEQ;

    float mx = -INFINITY;
#pragma unroll
    for (int j = 0; j < 32; ++j) {
      const int col = sub * 32 + j;
      float x = Sc[row * SLD + col];
      x = mrow[col] ? -INFINITY : x * scale;
      Sc[row * SLD + col] = x;
      mx = fmaxf(mx, x);
    }
#pragma unroll
    for (int m = 1; m < 16; m <<= 1) mx = fmaxf(mx, __shfl_xor(mx, m, 32));
    const bool any = (mx > -INFINITY);

    float sum = 0.f;
#pragma unroll
    for (int j = 0; j < 32; ++j) {
      const int col = sub * 32 + j;
      const float e = any ? __expf(Sc[row * SLD + col] - mx) : 0.f;
      Sc[row * SLD + col] = e;
      sum += e;
    }
#pragma unroll
    for (int m = 1; m < 16; m <<= 1) sum += __shfl_xor(sum, m, 32);
    const float inv = (sum > 0.f) ? (1.f / sum) : 0.f;   // all-masked row -> 0

#pragma unroll
    for (int j = 0; j < 32; ++j) {
      const int col = sub * 32 + j;
      Pbf[row * PLD + col] = (__bf16)(Sc[row * SLD + col] * inv);
    }
  }
  __syncthreads();

  // ---- P @ V: waves 0,1 each one 16x16 output tile over K=512 ----
  if (wave < 2) {
    const int nt = wave;
    v8f acc = {};
#pragma unroll
    for (int ks = 0; ks < 16; ++ks) {
      const v16bf ap = load_a_frag(&Pbf[l16 * PLD + ks * 32], half);
      const v16bf bv = load_b_frag(&Vts[(nt * 16 + l16) * PLD + ks * 32], half);
      acc = __builtin_amdgcn_wmma_f32_16x16x32_bf16(false, ap, false, bv,
                                                    (short)0, acc, false, false);
    }
    const int n = hh * HDIM + nt * 16 + l16;
#pragma unroll
    for (int r = 0; r < 8; ++r) {
      const int m = bb * SEQ + q0 + 8 * half + r;
      AOb[(size_t)m * DMODEL + n] = (__bf16)acc[r];
    }
  }
}

// ---------------------------------------------------------------------------
// h = LayerNorm(h + y) * g + b ; also writes bf16 copy for the next GEMMs.
// ---------------------------------------------------------------------------
__global__ __launch_bounds__(256)
void add_ln_kernel(float* __restrict__ h, __bf16* __restrict__ hb,
                   const float* __restrict__ y,
                   const float* __restrict__ g, const float* __restrict__ b) {
  __shared__ float red[256];
  const int row = blockIdx.x;
  const int d   = threadIdx.x;
  const size_t idx = (size_t)row * DMODEL + d;
  const float v = h[idx] + y[idx];

  red[d] = v;
  __syncthreads();
  for (int s = 128; s > 0; s >>= 1) {
    if (d < s) red[d] += red[d + s];
    __syncthreads();
  }
  const float mu = red[0] * (1.f / DMODEL);
  __syncthreads();

  const float c = v - mu;
  red[d] = c * c;
  __syncthreads();
  for (int s = 128; s > 0; s >>= 1) {
    if (d < s) red[d] += red[d + s];
    __syncthreads();
  }
  const float var = red[0] * (1.f / DMODEL);
  const float rs  = rsqrtf(var + 1e-5f);
  const float o   = c * rs * g[d] + b[d];
  h[idx]  = o;
  hb[idx] = (__bf16)o;
}

// ---------------------------------------------------------------------------
// h = x @ Win + bin (K = 5, scalar); writes fp32 + bf16.
// ---------------------------------------------------------------------------
__global__ __launch_bounds__(256)
void inproj_kernel(const float* __restrict__ x, const float* __restrict__ Win,
                   const float* __restrict__ binp,
                   float* __restrict__ h, __bf16* __restrict__ hb) {
  const int row = blockIdx.x;
  const int d   = threadIdx.x;
  float acc = binp[d];
#pragma unroll
  for (int f = 0; f < 5; ++f)
    acc += x[(size_t)row * 5 + f] * Win[f * DMODEL + d];
  const size_t idx = (size_t)row * DMODEL + d;
  h[idx]  = acc;
  hb[idx] = (__bf16)acc;
}

// ---------------------------------------------------------------------------
// One-time per call: Wt[n*K+k] = bf16(W[k*N+n])   (weights -> bf16, [N][K])
// ---------------------------------------------------------------------------
__global__ __launch_bounds__(256)
void wtrans_kernel(const float* __restrict__ W, __bf16* __restrict__ Wt,
                   int K, int N) {
  const int i = blockIdx.x * 256 + threadIdx.x;  // over K*N
  const int k = i / N, n = i % N;
  Wt[(size_t)n * K + k] = (__bf16)W[i];
}

// ---------------------------------------------------------------------------
extern "C" void kernel_launch(void* const* d_in, const int* in_sizes, int n_in,
                              void* d_out, int out_size, void* d_ws, size_t ws_size,
                              hipStream_t stream) {
  const float* x    = (const float*)d_in[0];
  const unsigned char* mask = (const unsigned char*)d_in[1];  // jnp bool, 1B/elem
  const float* Win  = (const float*)d_in[2];
  const float* binp = (const float*)d_in[3];
  const float* Wq   = (const float*)d_in[4];
  const float* bq   = (const float*)d_in[5];
  const float* Wk   = (const float*)d_in[6];
  const float* bk   = (const float*)d_in[7];
  const float* Wv   = (const float*)d_in[8];
  const float* bv   = (const float*)d_in[9];
  const float* Wo   = (const float*)d_in[10];
  const float* bo   = (const float*)d_in[11];
  const float* g1   = (const float*)d_in[12];
  const float* b1   = (const float*)d_in[13];
  const float* W1   = (const float*)d_in[14];
  const float* c1   = (const float*)d_in[15];
  const float* W2   = (const float*)d_in[16];
  const float* c2   = (const float*)d_in[17];
  const float* g2   = (const float*)d_in[18];
  const float* b2   = (const float*)d_in[19];

  const size_t HSZ = (size_t)MTOT * DMODEL;     // 4,194,304 elements
  float*  h   = (float*)d_ws;                   // fp32 hidden (persistent)
  float*  y   = h + HSZ;                        // fp32 temp (proj/FFN output)
  __bf16* hb  = (__bf16*)(h + 2 * HSZ);         // bf16 hidden
  __bf16* qb  = hb + HSZ;                       // ┐ q/k/v/ao bf16; region
  __bf16* kb  = qb + HSZ;                       // │ reused as the 16384x1024
  __bf16* vb  = kb + HSZ;                       // │ bf16 FFN hidden buffer
  __bf16* aob = vb + HSZ;                       // ┘
  __bf16* ffh = qb;                             // 4*HSZ == MTOT*FFDIM exactly
  __bf16* wt  = aob + HSZ;                      // transposed bf16 weights

  const size_t PW = (size_t)DMODEL * DMODEL;    // 65536
  const size_t LW = 4 * PW + 2 * (size_t)DMODEL * FFDIM;  // per-layer wt stride

  // ---- weights -> bf16 [N][K] (deterministic, every call) ----
  for (int l = 0; l < LAYERS; ++l) {
    __bf16* wl = wt + (size_t)l * LW;
    wtrans_kernel<<<PW / 256, 256, 0, stream>>>(Wq + l * PW, wl,           DMODEL, DMODEL);
    wtrans_kernel<<<PW / 256, 256, 0, stream>>>(Wk + l * PW, wl + PW,      DMODEL, DMODEL);
    wtrans_kernel<<<PW / 256, 256, 0, stream>>>(Wv + l * PW, wl + 2 * PW,  DMODEL, DMODEL);
    wtrans_kernel<<<PW / 256, 256, 0, stream>>>(Wo + l * PW, wl + 3 * PW,  DMODEL, DMODEL);
    wtrans_kernel<<<(DMODEL * FFDIM) / 256, 256, 0, stream>>>(
        W1 + (size_t)l * DMODEL * FFDIM, wl + 4 * PW, DMODEL, FFDIM);
    wtrans_kernel<<<(DMODEL * FFDIM) / 256, 256, 0, stream>>>(
        W2 + (size_t)l * FFDIM * DMODEL, wl + 4 * PW + (size_t)DMODEL * FFDIM,
        FFDIM, DMODEL);
  }

  inproj_kernel<<<MTOT, 256, 0, stream>>>(x, Win, binp, h, hb);

  const dim3 gP(DMODEL / BN, MTOT / BM);   // (4, 128)
  const dim3 gF(FFDIM / BN, MTOT / BM);    // (16, 128)
  const dim3 gA(SEQ / QC, NHEADS, BATCH);  // (32, 8, 32)

  for (int l = 0; l < LAYERS; ++l) {
    __bf16* wl = wt + (size_t)l * LW;
    const __bf16* wq_t = wl;
    const __bf16* wk_t = wl + PW;
    const __bf16* wv_t = wl + 2 * PW;
    const __bf16* wo_t = wl + 3 * PW;
    const __bf16* w1_t = wl + 4 * PW;                             // [FF][D]
    const __bf16* w2_t = wl + 4 * PW + (size_t)DMODEL * FFDIM;    // [D][FF]

    gemm_bf16_async<<<gP, 256, 0, stream>>>(hb, wq_t, bq + l * DMODEL, qb,
                                            MTOT, DMODEL, DMODEL, 0, 1);
    gemm_bf16_async<<<gP, 256, 0, stream>>>(hb, wk_t, bk + l * DMODEL, kb,
                                            MTOT, DMODEL, DMODEL, 0, 1);
    gemm_bf16_async<<<gP, 256, 0, stream>>>(hb, wv_t, bv + l * DMODEL, vb,
                                            MTOT, DMODEL, DMODEL, 0, 1);

    attn_kernel<<<gA, 256, 0, stream>>>(qb, kb, vb, mask, aob);

    gemm_bf16_async<<<gP, 256, 0, stream>>>(aob, wo_t, bo + l * DMODEL, y,
                                            MTOT, DMODEL, DMODEL, 0, 0);
    add_ln_kernel<<<MTOT, 256, 0, stream>>>(h, hb, y, g1 + l * DMODEL, b1 + l * DMODEL);

    gemm_bf16_async<<<gF, 256, 0, stream>>>(hb, w1_t, c1 + l * FFDIM, ffh,
                                            MTOT, FFDIM, DMODEL, 1, 1);
    gemm_bf16_async<<<gP, 256, 0, stream>>>(ffh, w2_t, c2 + l * DMODEL, y,
                                            MTOT, DMODEL, FFDIM, 0, 0);
    add_ln_kernel<<<MTOT, 256, 0, stream>>>(h, hb, y, g2 + l * DMODEL, b2 + l * DMODEL);
  }

  hipMemcpyAsync(d_out, h, HSZ * sizeof(float), hipMemcpyDeviceToDevice, stream);
}
